// GNN_32650341384571
// MI455X (gfx1250) — matrix operand order repaired and checked by reading, not compiled, not run on hardware
//
#include <hip/hip_runtime.h>

#define NNODES  50000
#define NEDGES  800000
#define NLAYERS 5

typedef __attribute__((ext_vector_type(16))) _Float16 v16h;
typedef __attribute__((ext_vector_type(8)))  _Float16 v8h;
typedef __attribute__((ext_vector_type(8)))  float    v8f;
typedef __attribute__((ext_vector_type(4)))  float    v4f;
typedef __attribute__((ext_vector_type(4)))  unsigned v4u;

#define ENC_NEG_INF 0x007FFFFFu   // order-preserving encoding of -inf
#define CAT_STRIDE  264           // 64-row f16 tile stride: 16B aligned, conflict-free
#define OLD_STRIDE  68            // f32 residual tile stride

// ---- order-preserving float<->uint for atomic segment-max ----
__device__ __forceinline__ unsigned encf(float f) {
  unsigned u = __float_as_uint(f);
  return (u & 0x80000000u) ? ~u : (u | 0x80000000u);
}
__device__ __forceinline__ float decf(unsigned k) {
  if (k == ENC_NEG_INF) return 0.0f;   // empty segment -> 0 (jraph semantics)
  return (k & 0x80000000u) ? __uint_as_float(k & 0x7FFFFFFFu)
                           : __uint_as_float(~k);
}
__device__ __forceinline__ v8h cvt8(v4f a, v4f b) {
  v8h r;
  r[0]=(_Float16)a.x; r[1]=(_Float16)a.y; r[2]=(_Float16)a.z; r[3]=(_Float16)a.w;
  r[4]=(_Float16)b.x; r[5]=(_Float16)b.y; r[6]=(_Float16)b.z; r[7]=(_Float16)b.w;
  return r;
}
__device__ __forceinline__ v8h cvt8u(v4u a, v4u b) {
  v8h r;
  r[0]=(_Float16)decf(a.x); r[1]=(_Float16)decf(a.y);
  r[2]=(_Float16)decf(a.z); r[3]=(_Float16)decf(a.w);
  r[4]=(_Float16)decf(b.x); r[5]=(_Float16)decf(b.y);
  r[6]=(_Float16)decf(b.z); r[7]=(_Float16)decf(b.w);
  return r;
}

// 16x256 @ 256x64 tile: 8 K-chunks x 4 N-tiles of v_wmma_f32_16x16x32_f16.
// catrow points at this lane's M-row (lane%16) in the LDS cat tile.
// packedW is fragment-major: ((kc*4+nt)*32 + lane)*16 contiguous f16.
__device__ __forceinline__ void wmma_tile(const _Float16* catrow,
                                          const _Float16* packedW,
                                          int lane, v8f acc[4]) {
  const int half = lane >> 4;
#pragma unroll
  for (int kc = 0; kc < 8; ++kc) {
    const v8h a0 = *(const v8h*)(catrow + kc * 32 + half * 8);
    const v8h a1 = *(const v8h*)(catrow + kc * 32 + half * 8 + 16);
    v16h a = __builtin_shufflevector(a0, a1, 0,1,2,3,4,5,6,7,8,9,10,11,12,13,14,15);
#pragma unroll
    for (int nt = 0; nt < 4; ++nt) {
      const _Float16* bp = packedW + (size_t)(((kc * 4 + nt) * 32 + lane) * 16);
      const v8h b0 = *(const v8h*)bp;
      const v8h b1 = *(const v8h*)(bp + 8);
      v16h b = __builtin_shufflevector(b0, b1, 0,1,2,3,4,5,6,7,8,9,10,11,12,13,14,15);
      acc[nt] = __builtin_amdgcn_wmma_f32_16x16x32_f16(
          false, a, false, b, (short)0, acc[nt], false, false);
    }
  }
}

// ---- pack We / Wn (f32 [L,256,64]) into WMMA B-fragment f16 layout ----
__global__ __launch_bounds__(256)
void pack_weights_kernel(const float* __restrict__ We,
                         const float* __restrict__ Wn,
                         _Float16* __restrict__ packed) {
  int idx = blockIdx.x * 256 + threadIdx.x;   // 10 * 16384 total
  int jp   = idx & 15;
  int lane = (idx >> 4) & 31;
  int frag = (idx >> 9) & 31;                 // kc*4 + nt
  int lm   = idx >> 14;                       // layer*2 + which
  int kc = frag >> 2, nt = frag & 3;
  int layer = lm >> 1, which = lm & 1;
  int K = kc * 32 + ((jp >> 3) * 16) + ((lane >> 4) * 8) + (jp & 7);  // B layout: lane half selects 8-wide K group in each 16-half
  int Nn = nt * 16 + (lane & 15);
  const float* src = which ? Wn : We;
  packed[idx] = (_Float16)src[((size_t)layer * 256 + K) * 64 + Nn];
}

__global__ void init_agg_kernel(unsigned* __restrict__ p, size_t n) {
  size_t i = (size_t)blockIdx.x * 256 + threadIdx.x;
  if (i < n) p[i] = ENC_NEG_INF;
}

// ---- embeddings ----
__global__ __launch_bounds__(256)
void edge_emb_kernel(const float* __restrict__ edges, const float* __restrict__ W,
                     const float* __restrict__ b, float* __restrict__ ebuf) {
  __shared__ float rows[4][16];
  int t = threadIdx.x, r0 = blockIdx.x * 4;
  if (t < 64) rows[t >> 4][t & 15] = edges[(size_t)r0 * 16 + t];
  __syncthreads();
  int mr = t >> 6, c = t & 63;
  float s = b[c];
#pragma unroll
  for (int k = 0; k < 16; ++k) s += rows[mr][k] * W[k * 64 + c];
  ebuf[(size_t)(r0 + mr) * 64 + c] = s;
}

__global__ __launch_bounds__(256)
void node_emb_kernel(const float* __restrict__ nodes, const float* __restrict__ W,
                     const float* __restrict__ b, float* __restrict__ vbuf) {
  __shared__ float rows[4][32];
  int t = threadIdx.x, r0 = blockIdx.x * 4;
  if (t < 128) rows[t >> 5][t & 31] = nodes[(size_t)r0 * 32 + t];
  __syncthreads();
  int mr = t >> 6, c = t & 63;
  float s = b[c];
#pragma unroll
  for (int k = 0; k < 32; ++k) s += rows[mr][k] * W[k * 64 + c];
  vbuf[(size_t)(r0 + mr) * 64 + c] = s;
}

__global__ __launch_bounds__(64)
void glob_emb_kernel(const float* __restrict__ g, const float* __restrict__ W,
                     const float* __restrict__ b, float* __restrict__ gbuf) {
  int c = threadIdx.x;
  float s = b[c];
#pragma unroll
  for (int k = 0; k < 8; ++k) s += g[k] * W[k * 64 + c];
  gbuf[c] = s;
}

// ---- fused edge update: gather -> WMMA MLP -> relu+residual -> seg-max scatter ----
__global__ __launch_bounds__(128)
void edge_update_kernel(float* __restrict__ ebuf, const float* __restrict__ vbuf,
                        const float* __restrict__ gvec,
                        const int* __restrict__ senders, const int* __restrict__ receivers,
                        const _Float16* __restrict__ packedW, const float* __restrict__ bias,
                        unsigned* __restrict__ sentbuf, unsigned* __restrict__ recvbuf,
                        unsigned* __restrict__ aggE) {
  __shared__ alignas(16) _Float16 cat[64][CAT_STRIDE];
  __shared__ alignas(16) float    eold[64][OLD_STRIDE];
  __shared__ int sidx[64], ridx[64];
  __shared__ unsigned colmax[64];
  const int t = threadIdx.x;
  const int e0 = blockIdx.x * 64;
  {
    int m = t >> 1, h = t & 1;
    int ge = e0 + m;
    const float* erow = ebuf + (size_t)ge * 64 + h * 32;
    const float* srow = vbuf + (size_t)senders[ge] * 64 + h * 32;
    const float* rrow = vbuf + (size_t)receivers[ge] * 64 + h * 32;
    const float* grow = gvec + h * 32;
#pragma unroll
    for (int j = 0; j < 32; j += 8) {
      v4f e1 = *(const v4f*)(erow + j), e2 = *(const v4f*)(erow + j + 4);
      *(v4f*)&eold[m][h * 32 + j]     = e1;
      *(v4f*)&eold[m][h * 32 + j + 4] = e2;
      *(v8h*)&cat[m][h * 32 + j] = cvt8(e1, e2);
      v4f s1 = *(const v4f*)(srow + j), s2 = *(const v4f*)(srow + j + 4);
      *(v8h*)&cat[m][64 + h * 32 + j] = cvt8(s1, s2);
      v4f r1 = *(const v4f*)(rrow + j), r2 = *(const v4f*)(rrow + j + 4);
      *(v8h*)&cat[m][128 + h * 32 + j] = cvt8(r1, r2);
      v4f g1 = *(const v4f*)(grow + j), g2 = *(const v4f*)(grow + j + 4);
      *(v8h*)&cat[m][192 + h * 32 + j] = cvt8(g1, g2);
    }
    if (t < 64) { sidx[t] = senders[e0 + t]; ridx[t] = receivers[e0 + t]; colmax[t] = ENC_NEG_INF; }
  }
  __syncthreads();

  const int lane = t & 31, wave = t >> 5;
  const int half = lane >> 4, r = lane & 15;
  v8f acc[4] = {v8f{}, v8f{}, v8f{}, v8f{}};
  wmma_tile(&cat[wave * 16 + r][0], packedW, lane, acc);

#pragma unroll
  for (int nt = 0; nt < 4; ++nt) {
    int n = nt * 16 + r;
    float bn = bias[n];
#pragma unroll
    for (int i = 0; i < 8; ++i) {
      int M = i + 8 * half;
      int lrow = wave * 16 + M;
      float x = acc[nt][i] + bn;
      x = x > 0.0f ? x : 0.0f;
      float out = eold[lrow][n] + x;
      ebuf[(size_t)(e0 + lrow) * 64 + n] = out;
      unsigned enc = encf(out);
      atomicMax(&sentbuf[(size_t)sidx[lrow] * 64 + n], enc);
      atomicMax(&recvbuf[(size_t)ridx[lrow] * 64 + n], enc);
      atomicMax(&colmax[n], enc);
    }
  }
  __syncthreads();
  if (t < 64) atomicMax(&aggE[t], colmax[t]);
}

// ---- fused node update ----
__global__ __launch_bounds__(128)
void node_update_kernel(float* __restrict__ vbuf,
                        const unsigned* __restrict__ sentbuf,
                        const unsigned* __restrict__ recvbuf,
                        const float* __restrict__ gvec,
                        const _Float16* __restrict__ packedW, const float* __restrict__ bias,
                        unsigned* __restrict__ aggV) {
  __shared__ alignas(16) _Float16 cat[64][CAT_STRIDE];
  __shared__ alignas(16) float    vold[64][OLD_STRIDE];
  __shared__ unsigned colmax[64];
  const int t = threadIdx.x;
  const int n0 = blockIdx.x * 64;
  {
    int m = t >> 1, h = t & 1;
    int gr = n0 + m;
    int grc = gr < NNODES ? gr : NNODES - 1;
    const float*    vrow = vbuf    + (size_t)grc * 64 + h * 32;
    const unsigned* srow = sentbuf + (size_t)grc * 64 + h * 32;
    const unsigned* rrow = recvbuf + (size_t)grc * 64 + h * 32;
    const float*    grow = gvec + h * 32;
#pragma unroll
    for (int j = 0; j < 32; j += 8) {
      v4f v1 = *(const v4f*)(vrow + j), v2 = *(const v4f*)(vrow + j + 4);
      *(v4f*)&vold[m][h * 32 + j]     = v1;
      *(v4f*)&vold[m][h * 32 + j + 4] = v2;
      *(v8h*)&cat[m][h * 32 + j] = cvt8(v1, v2);
      v4u s1 = *(const v4u*)(srow + j), s2 = *(const v4u*)(srow + j + 4);
      *(v8h*)&cat[m][64 + h * 32 + j] = cvt8u(s1, s2);
      v4u r1 = *(const v4u*)(rrow + j), r2 = *(const v4u*)(rrow + j + 4);
      *(v8h*)&cat[m][128 + h * 32 + j] = cvt8u(r1, r2);
      v4f g1 = *(const v4f*)(grow + j), g2 = *(const v4f*)(grow + j + 4);
      *(v8h*)&cat[m][192 + h * 32 + j] = cvt8(g1, g2);
    }
    if (t < 64) colmax[t] = ENC_NEG_INF;
  }
  __syncthreads();

  const int lane = t & 31, wave = t >> 5;
  const int half = lane >> 4, r = lane & 15;
  v8f acc[4] = {v8f{}, v8f{}, v8f{}, v8f{}};
  wmma_tile(&cat[wave * 16 + r][0], packedW, lane, acc);

#pragma unroll
  for (int nt = 0; nt < 4; ++nt) {
    int n = nt * 16 + r;
    float bn = bias[n];
#pragma unroll
    for (int i = 0; i < 8; ++i) {
      int M = i + 8 * half;
      int lrow = wave * 16 + M;
      int row = n0 + lrow;
      if (row < NNODES) {
        float x = acc[nt][i] + bn;
        x = x > 0.0f ? x : 0.0f;
        float out = vold[lrow][n] + x;
        vbuf[(size_t)row * 64 + n] = out;
        atomicMax(&colmax[n], encf(out));
      }
    }
  }
  __syncthreads();
  if (t < 64) atomicMax(&aggV[t], colmax[t]);
}

// ---- global update: g += relu([max(v), max(e), g] @ Wg + bg) ----
__global__ __launch_bounds__(64)
void glob_update_kernel(const unsigned* __restrict__ aggV, const unsigned* __restrict__ aggE,
                        float* __restrict__ gbuf, const float* __restrict__ Wg,
                        const float* __restrict__ bg) {
  __shared__ float catg[192];
  int t = threadIdx.x;
  catg[t]       = decf(aggV[t]);
  catg[64 + t]  = decf(aggE[t]);
  catg[128 + t] = gbuf[t];
  __syncthreads();
  float s = bg[t];
#pragma unroll 4
  for (int k = 0; k < 192; ++k) s += catg[k] * Wg[k * 64 + t];
  float rl = s > 0.0f ? s : 0.0f;
  gbuf[t] = catg[128 + t] + rl;
}

// ---- decoder: e_dec = e @ W_dec + b ----
__global__ __launch_bounds__(256)
void decode_kernel(const float* __restrict__ ebuf, const float* __restrict__ Wd,
                   const float* __restrict__ bd, float* __restrict__ out) {
  __shared__ float w[64];
  int t = threadIdx.x;
  if (t < 64) w[t] = Wd[t];
  __syncthreads();
  size_t row = (size_t)blockIdx.x * 256 + t;
  const float* er = ebuf + row * 64;
  float s = bd[0];
#pragma unroll
  for (int k = 0; k < 64; k += 4) {
    v4f x = *(const v4f*)(er + k);
    s += x.x * w[k] + x.y * w[k + 1] + x.z * w[k + 2] + x.w * w[k + 3];
  }
  out[row] = s;
}

extern "C" void kernel_launch(void* const* d_in, const int* in_sizes, int n_in,
                              void* d_out, int out_size, void* d_ws, size_t ws_size,
                              hipStream_t stream) {
  const float* nodes    = (const float*)d_in[0];
  const float* edges    = (const float*)d_in[1];
  const float* globals_ = (const float*)d_in[2];
  const int*   senders  = (const int*)d_in[3];
  const int*   receivers= (const int*)d_in[4];
  const float* W_e_emb  = (const float*)d_in[5];
  const float* b_e_emb  = (const float*)d_in[6];
  const float* W_n_emb  = (const float*)d_in[7];
  const float* b_n_emb  = (const float*)d_in[8];
  const float* W_g_emb  = (const float*)d_in[9];
  const float* b_g_emb  = (const float*)d_in[10];
  const float* We       = (const float*)d_in[11];
  const float* be       = (const float*)d_in[12];
  const float* Wn       = (const float*)d_in[13];
  const float* bn       = (const float*)d_in[14];
  const float* Wg       = (const float*)d_in[15];
  const float* bg       = (const float*)d_in[16];
  const float* W_dec    = (const float*)d_in[17];
  const float* b_dec    = (const float*)d_in[18];

  char* ws = (char*)d_ws;
  float* ebuf = (float*)ws;          ws += (size_t)NEDGES * 64 * 4;
  float* vbuf = (float*)ws;          ws += (size_t)NNODES * 64 * 4;
  unsigned* sentbuf = (unsigned*)ws; ws += (size_t)NNODES * 64 * 4;
  unsigned* recvbuf = (unsigned*)ws; ws += (size_t)NNODES * 64 * 4;
  unsigned* aggE = (unsigned*)ws;    ws += 256;
  unsigned* aggV = (unsigned*)ws;    ws += 256;
  float* gbuf = (float*)ws;          ws += 256;
  _Float16* packedW = (_Float16*)ws; ws += (size_t)NLAYERS * 2 * 16384 * 2;

  // weight packing + embeddings
  pack_weights_kernel<<<(NLAYERS * 2 * 16384) / 256, 256, 0, stream>>>(We, Wn, packedW);
  edge_emb_kernel<<<NEDGES / 4, 256, 0, stream>>>(edges, W_e_emb, b_e_emb, ebuf);
  node_emb_kernel<<<NNODES / 4, 256, 0, stream>>>(nodes, W_n_emb, b_n_emb, vbuf);
  glob_emb_kernel<<<1, 64, 0, stream>>>(globals_, W_g_emb, b_g_emb, gbuf);

  const size_t agg_words = (size_t)NNODES * 64 * 2 + 128;  // sent+recv+aggE+aggV (contiguous)
  for (int l = 0; l < NLAYERS; ++l) {
    init_agg_kernel<<<(int)((agg_words + 255) / 256), 256, 0, stream>>>(sentbuf, agg_words);
    edge_update_kernel<<<NEDGES / 64, 128, 0, stream>>>(
        ebuf, vbuf, gbuf, senders, receivers,
        packedW + (size_t)(l * 2) * 16384, be + l * 64, sentbuf, recvbuf, aggE);
    node_update_kernel<<<(NNODES + 63) / 64, 128, 0, stream>>>(
        vbuf, sentbuf, recvbuf, gbuf,
        packedW + (size_t)(l * 2 + 1) * 16384, bn + l * 64, aggV);
    glob_update_kernel<<<1, 64, 0, stream>>>(aggV, aggE, gbuf,
                                             Wg + (size_t)l * 192 * 64, bg + l * 64);
  }

  float* out = (float*)d_out;
  decode_kernel<<<NEDGES / 256, 256, 0, stream>>>(ebuf, W_dec, b_dec, out + (size_t)NNODES * 64);
  hipMemcpyAsync(out, vbuf, (size_t)NNODES * 64 * 4, hipMemcpyDeviceToDevice, stream);
  hipMemcpyAsync(out + (size_t)NNODES * 64 + NEDGES, gbuf, 64 * 4,
                 hipMemcpyDeviceToDevice, stream);
}
